// CriticCentral_offload_81509889344051
// MI455X (gfx1250) — matrix operand order repaired and checked by reading, not compiled, not run on hardware
//
#include <hip/hip_runtime.h>
#include <hip/hip_bf16.h>

typedef __attribute__((ext_vector_type(16))) _Float16 v16h;
typedef __attribute__((ext_vector_type(8)))  _Float16 v8h;
typedef __attribute__((ext_vector_type(8)))  float    v8f;

#define SC_STRIDE 130   // f16 scores row stride (128 + pad) -> conflict-free row/col walks

__global__ __launch_bounds__(256) void critic_central_kernel(
    const float* __restrict__ state,
    const int* __restrict__ p_nrt, const int* __restrict__ p_nob,
    const int* __restrict__ p_nrs, const int* __restrict__ p_ntg,
    const float* __restrict__ W_RT, const float* __restrict__ b_RT,
    const float* __restrict__ W_OB, const float* __restrict__ b_OB,
    const float* __restrict__ W_RS, const float* __restrict__ b_RS,
    const float* __restrict__ W_TG, const float* __restrict__ b_TG,
    const float* __restrict__ W1, const float* __restrict__ b1,
    const float* __restrict__ W2, const float* __restrict__ b2,
    const float* __restrict__ W3, const float* __restrict__ b3,
    float* __restrict__ out)
{
  // ---- LDS (~62.8 KB total; up to 5 WGs/WGP on 320KB) ----
  __shared__ __align__(16) _Float16 Qs[128 * 16];        // 4 KB  (rows padded to 16 halves)
  __shared__ __align__(16) _Float16 Ks[128 * 16];        // 4 KB
  __shared__ __align__(16) _Float16 Kz[16];              // 32 B zero stub for lanes 16..31
  __shared__ __align__(16) _Float16 Sc[128 * SC_STRIDE]; // 33.3 KB scores / exp values
  __shared__ __align__(16) float    Vs[128 * 32];        // 16 KB (dv padded to 32)
  __shared__ float rowmax[128];
  __shared__ float rinv[128];
  __shared__ float part[256];
  __shared__ float meanA[128];
  __shared__ float hvec[64];
  __shared__ float act1[256];
  __shared__ float act2[256];

  const int t    = threadIdx.x;
  const int lane = t & 31;
  const int wave = t >> 5;
  const int b    = blockIdx.x;

  if (t < 16) Kz[t] = (_Float16)0.0f;   // visible after Phase A's __syncthreads()

  const int nrt = p_nrt[0], nob = p_nob[0], nrs = p_nrs[0];
  const int offs[4] = {0, nrt, nrt + nob, nrt + nob + nrs};
  const float* Wp[4] = {W_RT, W_OB, W_RS, W_TG};
  const float* bp[4] = {b_RT, b_OB, b_RS, b_TG};

  #pragma unroll
  for (int br = 0; br < 4; ++br) {
    const int   dq     = (br == 3) ? 16 : 8;
    const int   dk     = dq;
    const int   dv     = 2 * dq;
    const int   d3     = dq + dk + dv;                  // 32 or 64
    const float qscale = (br == 3) ? 0.25f : 0.35355339059327373f; // 1/sqrt(dk)
    const int   hoff   = (br == 3) ? 48 : br * 16;      // concat offset in hvec
    const float* W    = Wp[br];
    const float* bias = bp[br];
    const int    off  = offs[br];

    // ---- Phase A: P = obs @ W + b ; scatter into Q(f16,prescaled), K(f16), V(f32)
    {
      const int row  = t & 127;
      const int half = t >> 7;
      const float* sp = state + ((size_t)b * 512 + (size_t)(off + row)) * 3;
      const float x0 = sp[0];
      const float x1 = sp[1];
      const int jcnt = d3 >> 1;                     // compile-time per branch
      const int j0   = half * jcnt;
      #pragma unroll
      for (int jj = 0; jj < jcnt; ++jj) {
        const int j = j0 + jj;
        float p = fmaf(x0, W[j], fmaf(x1, W[d3 + j], bias[j]));
        if (j < dq)           Qs[row * 16 + j]             = (_Float16)(p * qscale);
        else if (j < dq + dk) Ks[row * 16 + (j - dq)]      = (_Float16)p;
        else                  Vs[row * 32 + (j - dq - dk)] = p;
      }
      if (half == 0) {                              // zero-pad K-dim to 16 halves
        #pragma unroll
        for (int j = 0; j < 8; ++j) {
          if (dq + j < 16) {
            Qs[row * 16 + dq + j] = (_Float16)0.0f;
            Ks[row * 16 + dq + j] = (_Float16)0.0f;
          }
        }
      }
    }
    __syncthreads();

    // ---- Phase B: scores = (Q/sqrt(dk)) @ K^T via v_wmma_f32_16x16x32_f16
    // A 16x32 f16 frag: lanes 0-15 K=0..15 contiguously; lanes 16-31 (K16..31) zero via Kz stub.
    // B 32x16 f16 frag: lane n in 0..15 holds K=0..15 contiguously; lanes 16-31 (K16..31)
    //                   must be zero -> those lanes load from the Kz zero stub (addr select).
    {
      const int tm    = wave;              // 8 waves -> 8 tile rows
      const int lrow  = lane & 15;
      const int koff  = (lane >> 4) * 8;
      const int mbase = tm * 16 + (lane >> 4) * 8;
      v8h z8 = {};
      const _Float16* aptr = (lane < 16) ? &Qs[(tm * 16 + lrow) * 16] : &Kz[0];
      v16h afrag = *(const v16h*)aptr;
      #pragma unroll
      for (int tn = 0; tn < 8; ++tn) {
        const _Float16* bptr = (lane < 16) ? &Ks[(tn * 16 + lrow) * 16] : &Kz[0];
        v16h bfrag = *(const v16h*)bptr;
        v8f  c = {};
        v8f  d = __builtin_amdgcn_wmma_f32_16x16x32_f16(
                     false, afrag, false, bfrag, (short)0, c, false, false);
        const int n = tn * 16 + lrow;
        #pragma unroll
        for (int r = 0; r < 8; ++r)
          Sc[(mbase + r) * SC_STRIDE + n] = (_Float16)d[r];
      }
    }
    __syncthreads();

    // ---- Phase C: per-row max, then exp written back in place + row sums
    {
      const int q  = t >> 1;
      const int c0 = (t & 1) * 64;
      float m = -3.0e38f;
      for (int ci = 0; ci < 64; ++ci)
        m = fmaxf(m, (float)Sc[q * SC_STRIDE + c0 + ci]);
      part[t] = m;
      __syncthreads();
      if (t < 128) rowmax[t] = fmaxf(part[2 * t], part[2 * t + 1]);
      __syncthreads();
      const float rmq = rowmax[q];
      float s = 0.0f;
      for (int ci = 0; ci < 64; ++ci) {
        float e = __expf((float)Sc[q * SC_STRIDE + c0 + ci] - rmq);
        Sc[q * SC_STRIDE + c0 + ci] = (_Float16)e;   // store exp for column pass
        s += e;
      }
      part[t] = s;
      __syncthreads();
      if (t < 128) rinv[t] = 1.0f / (part[2 * t] + part[2 * t + 1]);
      __syncthreads();
    }

    // ---- Phase D: meanA[k] = (1/128) * sum_q exp(s)*rinv[q]  (column walk, no conflicts)
    if (t < 128) {
      float s = 0.0f;
      for (int q = 0; q < 128; ++q)
        s += (float)Sc[q * SC_STRIDE + t] * rinv[q];
      meanA[t] = s * (1.0f / 128.0f);
    }
    __syncthreads();

    // ---- Phase E: h[hoff + j] = sum_k meanA[k] * V[k][j]
    if (t < dv) {
      float acc = 0.0f;
      for (int k = 0; k < 128; ++k) acc = fmaf(meanA[k], Vs[k * 32 + t], acc);
      hvec[hoff + t] = acc;
    }
    __syncthreads();
  }

  // ---- MLP: tanh(h@W1+b1) -> tanh(@W2+b2) -> @W3+b3
  {
    float acc = b1[t];
    for (int i = 0; i < 64; ++i) acc = fmaf(hvec[i], W1[i * 256 + t], acc);
    act1[t] = tanhf(acc);
  }
  __syncthreads();
  {
    float acc = b2[t];
    for (int i = 0; i < 256; ++i) acc = fmaf(act1[i], W2[i * 256 + t], acc);
    act2[t] = tanhf(acc);
  }
  __syncthreads();
  part[t] = act2[t] * W3[t];
  __syncthreads();
  for (int s = 128; s > 0; s >>= 1) {
    if (t < s) part[t] += part[t + s];
    __syncthreads();
  }
  if (t == 0) out[b] = part[0] + b3[0];
}

extern "C" void kernel_launch(void* const* d_in, const int* in_sizes, int n_in,
                              void* d_out, int out_size, void* d_ws, size_t ws_size,
                              hipStream_t stream) {
  const float* state = (const float*)d_in[0];
  const int*   n_rt  = (const int*)d_in[1];
  const int*   n_ob  = (const int*)d_in[2];
  const int*   n_rs  = (const int*)d_in[3];
  const int*   n_tg  = (const int*)d_in[4];
  const float* W_RT  = (const float*)d_in[5];
  const float* b_RT  = (const float*)d_in[6];
  const float* W_OB  = (const float*)d_in[7];
  const float* b_OB  = (const float*)d_in[8];
  const float* W_RS  = (const float*)d_in[9];
  const float* b_RS  = (const float*)d_in[10];
  const float* W_TG  = (const float*)d_in[11];
  const float* b_TG  = (const float*)d_in[12];
  const float* W1    = (const float*)d_in[13];
  const float* b1    = (const float*)d_in[14];
  const float* W2    = (const float*)d_in[15];
  const float* b2    = (const float*)d_in[16];
  const float* W3    = (const float*)d_in[17];
  const float* b3    = (const float*)d_in[18];
  float* out = (float*)d_out;

  const int B = in_sizes[0] / (512 * 3);   // 4096
  critic_central_kernel<<<B, 256, 0, stream>>>(
      state, n_rt, n_ob, n_rs, n_tg,
      W_RT, b_RT, W_OB, b_OB, W_RS, b_RS, W_TG, b_TG,
      W1, b1, W2, b2, W3, b3, out);
}
